// Encoder_55688545960036
// MI455X (gfx1250) — compile-verified, matrix operand order checked by reading, not run on hardware
//
#include <hip/hip_runtime.h>
#include <hip/hip_bf16.h>
#include <math.h>

typedef _Float16 v8h  __attribute__((ext_vector_type(8)));
typedef _Float16 v16h __attribute__((ext_vector_type(16)));
typedef float    v8f  __attribute__((ext_vector_type(8)));

#define BN_EPS 1e-5f

// ---------------------------------------------------------------------------
// Kernel 1: convert wb [16,1024,128] f32 -> f16
// ---------------------------------------------------------------------------
__global__ void k_cvt_wb(const float* __restrict__ wb, _Float16* __restrict__ wb16, int n) {
    int i = blockIdx.x * blockDim.x + threadIdx.x;
    if (i < n) wb16[i] = (_Float16)wb[i];
}

// ---------------------------------------------------------------------------
// Kernel 2: fused shared MLP: h = relu(bn(relu(bn(x@w1^T))@w2^T)) -> f16
// grid = 8192 (one block per point), block = 128
// ---------------------------------------------------------------------------
__global__ void k_mlp(const float* __restrict__ x,
                      const float* __restrict__ w1, const float* __restrict__ g1,
                      const float* __restrict__ b1, const float* __restrict__ m1,
                      const float* __restrict__ v1,
                      const float* __restrict__ w2, const float* __restrict__ g2,
                      const float* __restrict__ b2, const float* __restrict__ m2,
                      const float* __restrict__ v2,
                      _Float16* __restrict__ h16) {
    __shared__ float h1[64];
    int p = blockIdx.x;          // b*2048 + n
    int t = threadIdx.x;         // 0..127
    float x0 = x[p * 3 + 0], x1 = x[p * 3 + 1], x2 = x[p * 3 + 2];
    if (t < 64) {
        float s = x0 * w1[t * 3 + 0] + x1 * w1[t * 3 + 1] + x2 * w1[t * 3 + 2];
        s = (s - m1[t]) * (g1[t] * rsqrtf(v1[t] + BN_EPS)) + b1[t];
        h1[t] = fmaxf(s, 0.0f);
    }
    __syncthreads();
    const float* w2r = w2 + t * 64;
    float s = 0.0f;
#pragma unroll 8
    for (int c = 0; c < 64; ++c) s += h1[c] * w2r[c];
    s = (s - m2[t]) * (g2[t] * rsqrtf(v2[t] + BN_EPS)) + b2[t];
    s = fmaxf(s, 0.0f);
    h16[p * 128 + t] = (_Float16)s;
}

// ---------------------------------------------------------------------------
// Kernel 3: heavy GEMM + running max over N + fused BN (BN scale > 0 so
// max commutes with BN).  One wave per (b,k,16-wide o tile).
// grid = (16 o-chunks, 16 k, 4 b), block = 128 (4 waves; wave -> o tile)
// ---------------------------------------------------------------------------
__device__ inline v16h load_frag16(const _Float16* p) {
    v8h lo = *(const v8h*)(p);
    v8h hi = *(const v8h*)(p + 16);
    return __builtin_shufflevector(lo, hi, 0, 1, 2, 3, 4, 5, 6, 7,
                                   8, 9, 10, 11, 12, 13, 14, 15);
}

__global__ void k_gemm_max(const _Float16* __restrict__ h16,
                           const _Float16* __restrict__ wb16,
                           const float* __restrict__ gb, const float* __restrict__ bb,
                           const float* __restrict__ mb, const float* __restrict__ vb,
                           float* __restrict__ feat) {
    int wave = threadIdx.x >> 5;
    int lane = threadIdx.x & 31;
    int otile = blockIdx.x * 4 + wave;        // 0..63
    int k = blockIdx.y;
    int b = blockIdx.z;

    int rc   = lane & 15;                     // A row / B column within tile
    int kofs = (lane >> 4) ? 8 : 0;           // K sub-chunk per ISA 16-bit layout

    const _Float16* hbase = h16 + (size_t)b * 2048 * 128;
    const _Float16* wbase = wb16 + ((size_t)k * 1024 + (size_t)otile * 16) * 128;

    // B fragments for the 4 K-steps (reused across all 128 N tiles)
    v16h Bf[4];
#pragma unroll
    for (int s = 0; s < 4; ++s)
        Bf[s] = load_frag16(wbase + rc * 128 + 32 * s + kofs);

    v8f mx;
#pragma unroll
    for (int j = 0; j < 8; ++j) mx[j] = -3.4e38f;

    for (int nt = 0; nt < 128; ++nt) {
        const _Float16* arow = hbase + (size_t)(nt * 16 + rc) * 128;
        v8f acc = {};
#pragma unroll
        for (int s = 0; s < 4; ++s) {
            v16h Af = load_frag16(arow + 32 * s + kofs);
            acc = __builtin_amdgcn_wmma_f32_16x16x32_f16(
                false, Af, false, Bf[s], (short)0, acc, false, false);
        }
#pragma unroll
        for (int j = 0; j < 8; ++j) mx[j] = fmaxf(mx[j], acc[j]);
    }

    // reduce over the 8 accumulator rows (M=0..7 / 8..15), then across halves
    float m = mx[0];
#pragma unroll
    for (int j = 1; j < 8; ++j) m = fmaxf(m, mx[j]);
    m = fmaxf(m, __shfl_xor(m, 16, 32));

    if (lane < 16) {
        int o = otile * 16 + lane;
        int ko = k * 1024 + o;
        float scale = gb[ko] * rsqrtf(vb[ko] + BN_EPS);
        feat[((size_t)b * 16 + k) * 1024 + o] = (m - mb[ko]) * scale + bb[ko];
    }
}

// ---------------------------------------------------------------------------
// Kernel 4: caps[b,i,e] = squash_e(feat[b,e,i])
// grid = 256 blocks (16 i each), block = 256 (16 i-groups x 16 e)
// ---------------------------------------------------------------------------
__global__ void k_caps(const float* __restrict__ feat, float* __restrict__ caps) {
    int t = threadIdx.x;
    int e = t & 15;
    int il = t >> 4;
    int b = blockIdx.x >> 6;
    int i = (blockIdx.x & 63) * 16 + il;
    float val = feat[((size_t)b * 16 + e) * 1024 + i];
    float sq = val * val;
#pragma unroll
    for (int m = 1; m < 16; m <<= 1) sq += __shfl_xor(sq, m, 32);
    float n = sqrtf(sq);
    caps[((size_t)b * 1024 + i) * 16 + e] = val * (n / (1.0f + sq));
}

// ---------------------------------------------------------------------------
// Kernel 5: u[b,o,i,v] = sum_e caps[b,i,e] * Wc[o,i,e,v]
// grid = 4096, block = 256 (8 waves; wave -> (o,i)), lane = v
// ---------------------------------------------------------------------------
__global__ void k_u(const float* __restrict__ caps, const float* __restrict__ Wc,
                    float* __restrict__ u) {
    int w = blockIdx.x * 8 + (threadIdx.x >> 5);   // 0..32767
    int o = w >> 10;
    int i = w & 1023;
    int lane = threadIdx.x & 31;

    const float* wc = Wc + (((size_t)o * 1024 + i) * 16) * 32;
    float a0 = 0.f, a1 = 0.f, a2 = 0.f, a3 = 0.f;
#pragma unroll
    for (int e = 0; e < 16; ++e) {
        float wv = wc[e * 32 + lane];
        a0 += caps[((size_t)0 * 1024 + i) * 16 + e] * wv;
        a1 += caps[((size_t)1 * 1024 + i) * 16 + e] * wv;
        a2 += caps[((size_t)2 * 1024 + i) * 16 + e] * wv;
        a3 += caps[((size_t)3 * 1024 + i) * 16 + e] * wv;
    }
    size_t base = ((size_t)o * 1024 + i) * 32 + lane;
    u[((size_t)0 * 32 * 1024) * 32 + base] = a0;
    u[((size_t)1 * 32 * 1024) * 32 + base] = a1;
    u[((size_t)2 * 32 * 1024) * 32 + base] = a2;
    u[((size_t)3 * 32 * 1024) * 32 + base] = a3;
}

// ---------------------------------------------------------------------------
// Routing: s[b,o,v] = sum_i c[b,o,i] * u[b,o,i,v]  (use_c==0 -> c = 1/32)
// grid = 128 (b*32+o), block = 256 (8 waves split i), lane = v
// ---------------------------------------------------------------------------
__global__ void kr_s(const float* __restrict__ u, const float* __restrict__ cbuf,
                     float* __restrict__ sbuf, int use_c) {
    __shared__ float red[8][32];
    int g = blockIdx.x;               // b*32 + o
    int wv = threadIdx.x >> 5;
    int lane = threadIdx.x & 31;
    const float* ub = u + (size_t)g * 1024 * 32;
    float acc = 0.f;
    for (int i = wv; i < 1024; i += 8) {
        float cv = use_c ? cbuf[(size_t)g * 1024 + i] : 0.03125f;
        acc += ub[(size_t)i * 32 + lane] * cv;
    }
    red[wv][lane] = acc;
    __syncthreads();
    if (wv == 0) {
        float s = 0.f;
#pragma unroll
        for (int j = 0; j < 8; ++j) s += red[j][lane];
        sbuf[(size_t)g * 32 + lane] = s;
    }
}

// ---------------------------------------------------------------------------
// Squash over v: out[b,o,v] = s * n/(1+n^2)
// grid = 32, block = 128 (4 waves; wave -> (b,o)), lane = v
// ---------------------------------------------------------------------------
__global__ void kr_squash(const float* __restrict__ sbuf, float* __restrict__ out) {
    int g = blockIdx.x * 4 + (threadIdx.x >> 5);
    int lane = threadIdx.x & 31;
    float sv = sbuf[(size_t)g * 32 + lane];
    float sq = sv * sv;
#pragma unroll
    for (int m = 1; m < 32; m <<= 1) sq += __shfl_xor(sq, m, 32);
    float n = sqrtf(sq);
    out[(size_t)g * 32 + lane] = sv * (n / (1.0f + sq));
}

// ---------------------------------------------------------------------------
// Agreement + b_log update + softmax over o -> c
// grid = 512, block = 256 (8 waves; wave -> (b,i)), lane = v
// ---------------------------------------------------------------------------
__global__ void kr_agree(const float* __restrict__ u, const float* __restrict__ abuf,
                         float* __restrict__ blog, float* __restrict__ cbuf,
                         int first) {
    int w = blockIdx.x * 8 + (threadIdx.x >> 5);   // 0..4095
    int b = w >> 10;
    int i = w & 1023;
    int lane = threadIdx.x & 31;

    float arr[32];
    float mx = -3.4e38f;
#pragma unroll 4
    for (int o = 0; o < 32; ++o) {
        size_t go = (size_t)(b * 32 + o);
        float d = abuf[go * 32 + lane] * u[(go * 1024 + i) * 32 + lane];
#pragma unroll
        for (int m = 1; m < 32; m <<= 1) d += __shfl_xor(d, m, 32);
        float bo = first ? d : (blog[go * 1024 + i] + d);
        if (lane == o) blog[go * 1024 + i] = bo;
        arr[o] = bo;
        mx = fmaxf(mx, bo);
    }
    float sum = 0.f;
#pragma unroll
    for (int o = 0; o < 32; ++o) { arr[o] = __expf(arr[o] - mx); sum += arr[o]; }
    float inv = 1.0f / sum;
#pragma unroll 4
    for (int o = 0; o < 32; ++o) {
        if (lane == o) cbuf[(size_t)(b * 32 + o) * 1024 + i] = arr[o] * inv;
    }
}

// ---------------------------------------------------------------------------
extern "C" void kernel_launch(void* const* d_in, const int* in_sizes, int n_in,
                              void* d_out, int out_size, void* d_ws, size_t ws_size,
                              hipStream_t stream) {
    const float* x  = (const float*)d_in[0];
    const float* w1 = (const float*)d_in[1];
    const float* g1 = (const float*)d_in[2];
    const float* b1 = (const float*)d_in[3];
    const float* m1 = (const float*)d_in[4];
    const float* v1 = (const float*)d_in[5];
    const float* w2 = (const float*)d_in[6];
    const float* g2 = (const float*)d_in[7];
    const float* b2 = (const float*)d_in[8];
    const float* m2 = (const float*)d_in[9];
    const float* v2 = (const float*)d_in[10];
    const float* wb = (const float*)d_in[11];
    const float* gb = (const float*)d_in[12];
    const float* bb = (const float*)d_in[13];
    const float* mb = (const float*)d_in[14];
    const float* vb = (const float*)d_in[15];
    const float* Wc = (const float*)d_in[16];
    float* out = (float*)d_out;

    char* w = (char*)d_ws;
    _Float16* h16  = (_Float16*)(w + 0);               // 2,097,152 B
    _Float16* wb16 = (_Float16*)(w + 2097152);         // 4,194,304 B
    float* feat = (float*)(w + 6291456);               //   262,144 B
    float* caps = (float*)(w + 6553600);               //   262,144 B
    float* u    = (float*)(w + 6815744);               // 16,777,216 B
    float* blog = (float*)(w + 23592960);              //   524,288 B
    float* cbuf = (float*)(w + 24117248);              //   524,288 B
    float* sbuf = (float*)(w + 24641536);              //    16,384 B
    float* abuf = (float*)(w + 24657920);              //    16,384 B

    // 1. precision conversion + fused MLP
    k_cvt_wb<<<(16 * 1024 * 128 + 255) / 256, 256, 0, stream>>>(wb, wb16, 16 * 1024 * 128);
    k_mlp<<<8192, 128, 0, stream>>>(x, w1, g1, b1, m1, v1, w2, g2, b2, m2, v2, h16);

    // 2. WMMA GEMM + fused max-over-N + BN
    k_gemm_max<<<dim3(16, 16, 4), 128, 0, stream>>>(h16, wb16, gb, bb, mb, vb, feat);

    // 3. capsule squash
    k_caps<<<256, 256, 0, stream>>>(feat, caps);

    // 4. capsule transform
    k_u<<<4096, 256, 0, stream>>>(caps, Wc, u);

    // 5. dynamic routing, T = 3
    kr_s<<<128, 256, 0, stream>>>(u, cbuf, sbuf, 0);        // c = 1/32
    kr_squash<<<32, 128, 0, stream>>>(sbuf, abuf);          // a0
    kr_agree<<<512, 256, 0, stream>>>(u, abuf, blog, cbuf, 1);

    kr_s<<<128, 256, 0, stream>>>(u, cbuf, sbuf, 1);
    kr_squash<<<32, 128, 0, stream>>>(sbuf, abuf);          // a1
    kr_agree<<<512, 256, 0, stream>>>(u, abuf, blog, cbuf, 0);

    kr_s<<<128, 256, 0, stream>>>(u, cbuf, sbuf, 1);
    kr_squash<<<32, 128, 0, stream>>>(sbuf, out);           // final squash -> d_out
}
